// ModernBertEncoderLayer_39015482917343
// MI455X (gfx1250) — compile-verified, matrix operand order checked by reading, not run on hardware
//
#include <hip/hip_runtime.h>

#define TOK   8192
#define SEQ   4096
#define DM    768
#define NH    12
#define DH    64
#define DI    1152
#define MASKV (-10000.0f)

// Async global->LDS path (ASYNCcnt-tracked) if this toolchain exposes it.
#if defined(__HIP_DEVICE_COMPILE__) && defined(__gfx1250__)
#  if __has_builtin(__builtin_amdgcn_global_load_async_to_lds_b128) && \
      __has_builtin(__builtin_amdgcn_s_wait_asynccnt)
#    define USE_ASYNC 1
#  endif
#endif
#ifndef USE_ASYNC
#  define USE_ASYNC 0
#endif

typedef __attribute__((ext_vector_type(16))) __bf16 v16bf;
typedef __attribute__((ext_vector_type(8)))  __bf16 bf16x8;
typedef __attribute__((ext_vector_type(8)))  float  v8f;

__device__ __forceinline__ __bf16 f2bf(float f) {
  unsigned u = __builtin_bit_cast(unsigned, f);
  u += 0x7FFFu + ((u >> 16) & 1u);               // round-to-nearest-even
  unsigned short s = (unsigned short)(u >> 16);
  return __builtin_bit_cast(__bf16, s);
}

__device__ __forceinline__ v8f fzero8() {
  v8f z;
#pragma unroll
  for (int i = 0; i < 8; ++i) z[i] = 0.0f;
  return z;
}

__device__ __forceinline__ bf16x8 bzero8() {
  bf16x8 z;
#pragma unroll
  for (int i = 0; i < 8; ++i) z[i] = f2bf(0.0f);
  return z;
}

__device__ __forceinline__ v8f wmma_bf16(v16bf a, v16bf b, v8f c) {
  // 8 args: (neg_a, A, neg_b, B, c_mod, C, reuse_a, reuse_b)
  return __builtin_amdgcn_wmma_f32_16x16x32_bf16(false, a, false, b, (short)0, c,
                                                 false, false);
}

#if USE_ASYNC
// 16B per-lane async copy global -> LDS, tracked by ASYNCcnt.
// Builtin expects (int4* global_src, int4* lds_dst, imm offset, imm cpol);
// generic pointers convert to the cuda_device/cuda_shared AS params in HIP.
typedef int v4i_ __attribute__((vector_size(16)));
__device__ __forceinline__ void async_cp16(const void* g, void* l) {
  __builtin_amdgcn_global_load_async_to_lds_b128((v4i_*)g, (v4i_*)l, 0, 0);
}
#endif

// 16x32 bf16 fragment from row-major LDS tile (leading dim ld, elements).
// ISA 16-bit A/B layout: lanes 0-15 hold K 0..7 / 16..23, lanes 16-31 hold
// K 8..15 / 24..31 -> two contiguous 16B chunks per lane = 2x ds_load_b128.
__device__ __forceinline__ v16bf ld_frag(const __bf16* base, int ld, int row0, int k0) {
  int lane = threadIdx.x & 31;
  const __bf16* p = base + (size_t)(row0 + (lane & 15)) * ld + k0 + ((lane >> 4) << 3);
  union { v16bf v; bf16x8 h[2]; } u;
  u.h[0] = *(const bf16x8*)p;
  u.h[1] = *(const bf16x8*)(p + 16);
  return u.v;
}

// ---------------------------------------------------------------------------
// Weight convert fp32 [K][N] -> bf16 transposed [N][K]
// ---------------------------------------------------------------------------
__global__ __launch_bounds__(256) void wtrans_kernel(const float* __restrict__ w,
                                                     __bf16* __restrict__ wt,
                                                     int K, int N) {
  size_t idx = (size_t)blockIdx.x * 256 + threadIdx.x;
  if (idx >= (size_t)K * N) return;
  int k = (int)(idx / N), n = (int)(idx % N);
  wt[(size_t)n * K + k] = f2bf(w[idx]);
}

// ---------------------------------------------------------------------------
// RMSNorm: fp32 row -> bf16 row
// ---------------------------------------------------------------------------
__global__ __launch_bounds__(256) void rmsnorm_kernel(const float* __restrict__ x,
                                                      const float* __restrict__ gamma,
                                                      __bf16* __restrict__ out) {
  __shared__ float red[9];
  int row = blockIdx.x;
  const float* xr = x + (size_t)row * DM;
  float ss = 0.0f;
  for (int i = threadIdx.x; i < DM; i += 256) { float v = xr[i]; ss += v * v; }
#pragma unroll
  for (int m = 1; m < 32; m <<= 1) ss += __shfl_xor(ss, m, 32);
  if ((threadIdx.x & 31) == 0) red[threadIdx.x >> 5] = ss;
  __syncthreads();
  if (threadIdx.x == 0) {
    float t = 0.0f;
#pragma unroll
    for (int i = 0; i < 8; ++i) t += red[i];
    red[8] = rsqrtf(t * (1.0f / DM) + 1e-5f);
  }
  __syncthreads();
  float inv = red[8];
  for (int i = threadIdx.x; i < DM; i += 256)
    out[(size_t)row * DM + i] = f2bf(xr[i] * inv * gamma[i]);
}

// ---------------------------------------------------------------------------
// Tiled bf16 GEMM, C[M,N] = A[M,K] * B^T (B stored [N][K]).
// Block tile 128x128, 8 waves (wave tile 32x64), BK=32, ping-pong LDS
// double buffering. Next tile is fetched (async-to-LDS, or global->VGPR with
// deferred ds_store) while the current tile feeds the WMMAs.
// MODE 0: QKV projection + fused RoPE, split-store to q/k/v bf16 buffers
// MODE 1: + fp32 residual, fp32 output
// MODE 2: dual-B (gate/up), fused gelu(gate)*up, bf16 output
// ---------------------------------------------------------------------------
template <int MODE>
__global__ __launch_bounds__(256) void gemm_kernel(
    const __bf16* __restrict__ A, const __bf16* __restrict__ B0,
    const __bf16* __restrict__ B1, const float* __restrict__ resid,
    float* __restrict__ outf, __bf16* __restrict__ outq, __bf16* __restrict__ outk,
    __bf16* __restrict__ outv, __bf16* __restrict__ outbf, int N, int K) {
  __shared__ __bf16 sA [2][128][40];   // pad LD=40: conflict-free b128 frags
  __shared__ __bf16 sB0[2][128][40];
  __shared__ __bf16 sB1[2][128][40];
  const int tid = threadIdx.x, lane = tid & 31, wave = tid >> 5;
  const int wm0 = (wave & 3) * 32, wn0 = (wave >> 2) * 64;
  const int n0 = blockIdx.x * 128, m0 = blockIdx.y * 128;
  const int lane15 = lane & 15, g = lane >> 4;

  // Per-thread staging: rows r0 and r0+64, one 8-elem chunk each, per stream.
  const int r0 = tid >> 2, c0 = (tid & 3) << 3;

  v8f acc[2][4], accU[2][4];
#pragma unroll
  for (int i = 0; i < 2; ++i)
#pragma unroll
    for (int j = 0; j < 4; ++j) { acc[i][j] = fzero8(); accU[i][j] = fzero8(); }

#if USE_ASYNC
  auto stage = [&](int kt, int buf) {
    async_cp16(&A [(size_t)(m0 + r0) * K + kt + c0],      &sA [buf][r0][c0]);
    async_cp16(&A [(size_t)(m0 + r0 + 64) * K + kt + c0], &sA [buf][r0 + 64][c0]);
    async_cp16(&B0[(size_t)(n0 + r0) * K + kt + c0],      &sB0[buf][r0][c0]);
    async_cp16(&B0[(size_t)(n0 + r0 + 64) * K + kt + c0], &sB0[buf][r0 + 64][c0]);
    if (MODE == 2) {
      async_cp16(&B1[(size_t)(n0 + r0) * K + kt + c0],      &sB1[buf][r0][c0]);
      async_cp16(&B1[(size_t)(n0 + r0 + 64) * K + kt + c0], &sB1[buf][r0 + 64][c0]);
    }
  };
  stage(0, 0);
  __builtin_amdgcn_s_wait_asynccnt(0);
#else
  bf16x8 ra[2], rb[2], rc[2];
  auto load_regs = [&](int kt) {
    ra[0] = *(const bf16x8*)&A [(size_t)(m0 + r0) * K + kt + c0];
    ra[1] = *(const bf16x8*)&A [(size_t)(m0 + r0 + 64) * K + kt + c0];
    rb[0] = *(const bf16x8*)&B0[(size_t)(n0 + r0) * K + kt + c0];
    rb[1] = *(const bf16x8*)&B0[(size_t)(n0 + r0 + 64) * K + kt + c0];
    if (MODE == 2) {
      rc[0] = *(const bf16x8*)&B1[(size_t)(n0 + r0) * K + kt + c0];
      rc[1] = *(const bf16x8*)&B1[(size_t)(n0 + r0 + 64) * K + kt + c0];
    }
  };
  auto store_lds = [&](int buf) {
    *(bf16x8*)&sA [buf][r0][c0]      = ra[0];
    *(bf16x8*)&sA [buf][r0 + 64][c0] = ra[1];
    *(bf16x8*)&sB0[buf][r0][c0]      = rb[0];
    *(bf16x8*)&sB0[buf][r0 + 64][c0] = rb[1];
    if (MODE == 2) {
      *(bf16x8*)&sB1[buf][r0][c0]      = rc[0];
      *(bf16x8*)&sB1[buf][r0 + 64][c0] = rc[1];
    }
  };
  load_regs(0);
  store_lds(0);
#endif
  __syncthreads();

  const int nk = K >> 5;
  for (int t = 0; t < nk; ++t) {
    const int cur = t & 1, nxt = cur ^ 1;
    const bool more = (t + 1) < nk;
#if USE_ASYNC
    if (more) stage((t + 1) << 5, nxt);     // fire-and-forget next tile
#else
    if (more) load_regs((t + 1) << 5);      // issue global loads early
#endif
    v16bf af0 = ld_frag(&sA[cur][0][0], 40, wm0, 0);
    v16bf af1 = ld_frag(&sA[cur][0][0], 40, wm0 + 16, 0);
#pragma unroll
    for (int j = 0; j < 4; ++j) {
      v16bf b = ld_frag(&sB0[cur][0][0], 40, wn0 + 16 * j, 0);
      acc[0][j] = wmma_bf16(af0, b, acc[0][j]);
      acc[1][j] = wmma_bf16(af1, b, acc[1][j]);
      if (MODE == 2) {
        v16bf b1 = ld_frag(&sB1[cur][0][0], 40, wn0 + 16 * j, 0);
        accU[0][j] = wmma_bf16(af0, b1, accU[0][j]);
        accU[1][j] = wmma_bf16(af1, b1, accU[1][j]);
      }
    }
#if USE_ASYNC
    __builtin_amdgcn_s_wait_asynccnt(0);
#else
    if (more) store_lds(nxt);
#endif
    __syncthreads();
  }

  if (MODE == 0) {
    const int section = n0 / DM;  // 0=q 1=k 2=v (768 % 128 == 0)
    __bf16* dst = (section == 0) ? outq : (section == 1) ? outk : outv;
#pragma unroll
    for (int i = 0; i < 2; ++i)
#pragma unroll
      for (int r = 0; r < 8; ++r) {
        int m = m0 + wm0 + 16 * i + r + 8 * g;
        float pos = (float)(m & (SEQ - 1));
#pragma unroll
        for (int jj = 0; jj < 2; ++jj) {
          int ncol = n0 + wn0 + 16 * jj + lane15;
          int d = (wn0 + 16 * jj + lane15) & 63;  // 0..31; pair at d+32 is jj+2
          float xa = acc[i][jj][r], xb = acc[i][jj + 2][r];
          if (section < 2) {                       // RoPE on q and k
            float inv = __expf(-0.2878231366f * (float)d);  // 10000^(-d/32)
            float fr = pos * inv, sn, cs;
            __sincosf(fr, &sn, &cs);
            float ra_ = xa * cs - xb * sn;
            float rb_ = xb * cs + xa * sn;
            xa = ra_; xb = rb_;
          }
          int c = ncol - section * DM;
          dst[(size_t)m * DM + c]      = f2bf(xa);
          dst[(size_t)m * DM + c + 32] = f2bf(xb);
        }
      }
  } else if (MODE == 1) {
#pragma unroll
    for (int i = 0; i < 2; ++i)
#pragma unroll
      for (int r = 0; r < 8; ++r) {
        int m = m0 + wm0 + 16 * i + r + 8 * g;
#pragma unroll
        for (int j = 0; j < 4; ++j) {
          int ncol = n0 + wn0 + 16 * j + lane15;
          size_t idx = (size_t)m * N + ncol;
          outf[idx] = acc[i][j][r] + resid[idx];
        }
      }
  } else {
#pragma unroll
    for (int i = 0; i < 2; ++i)
#pragma unroll
      for (int r = 0; r < 8; ++r) {
        int m = m0 + wm0 + 16 * i + r + 8 * g;
#pragma unroll
        for (int j = 0; j < 4; ++j) {
          int ncol = n0 + wn0 + 16 * j + lane15;
          float gt = acc[i][j][r], up = accU[i][j][r];
          float t2 = 0.7978845608f * (gt + 0.044715f * gt * gt * gt);
          float gl = 0.5f * gt * (1.0f + tanhf(t2));
          outbf[(size_t)m * N + ncol] = f2bf(gl * up);
        }
      }
  }
}

// ---------------------------------------------------------------------------
// Sliding-window attention: one block per (batch, head, 128-query strip).
// Keys span [q0-64, q0+191] = 256 keys. QK^T and P*V via bf16 WMMA.
// ---------------------------------------------------------------------------
__global__ __launch_bounds__(256) void attn_kernel(
    const __bf16* __restrict__ qb, const __bf16* __restrict__ kb,
    const __bf16* __restrict__ vb, const int* __restrict__ pmask,
    __bf16* __restrict__ ob) {
  extern __shared__ char smem[];
  __bf16* sQ  = (__bf16*)smem;          // [128][72]
  __bf16* sK  = sQ + 128 * 72;          // [256][72]
  __bf16* sVt = sK + 256 * 72;          // [64][264]  (V transposed: [d][key])
  __bf16* sP  = sVt + 64 * 264;         // [128][264]
  const int tid = threadIdx.x, lane = tid & 31, wave = tid >> 5;
  const int lane15 = lane & 15, g = lane >> 4;
  const int nq = SEQ / 128;
  const int q0 = (blockIdx.x % nq) * 128;
  const int h  = (blockIdx.x / nq) % NH;
  const int b  = blockIdx.x / (nq * NH);
  const size_t base = (size_t)b * SEQ * DM + (size_t)h * DH;
  const int k0 = q0 - 64;

  for (int i = tid; i < 128 * 8; i += 256) {
    int row = i >> 3, c = (i & 7) << 3;
    *(bf16x8*)&sQ[row * 72 + c] = *(const bf16x8*)&qb[base + (size_t)(q0 + row) * DM + c];
  }
  for (int i = tid; i < 256 * 8; i += 256) {
    int row = i >> 3, c = (i & 7) << 3;
    int key = k0 + row;
    bf16x8 kv = bzero8(), vv = bzero8();
    if (key >= 0 && key < SEQ) {
      kv = *(const bf16x8*)&kb[base + (size_t)key * DM + c];
      vv = *(const bf16x8*)&vb[base + (size_t)key * DM + c];
    }
    *(bf16x8*)&sK[row * 72 + c] = kv;
#pragma unroll
    for (int e = 0; e < 8; ++e) sVt[(c + e) * 264 + row] = vv[e];
  }
  __syncthreads();

  // scores: 16 query rows per wave x 256 keys
  v8f acc[16];
#pragma unroll
  for (int t = 0; t < 16; ++t) acc[t] = fzero8();
#pragma unroll
  for (int kk = 0; kk < DH; kk += 32) {
    v16bf aq = ld_frag(sQ, 72, wave * 16, kk);
#pragma unroll
    for (int t = 0; t < 16; ++t) {
      v16bf bk = ld_frag(sK, 72, 16 * t, kk);
      acc[t] = wmma_bf16(aq, bk, acc[t]);
    }
  }

  float padd[16]; int okk[16];
#pragma unroll
  for (int t = 0; t < 16; ++t) {
    int key = k0 + 16 * t + lane15;
    okk[t] = (key >= 0 && key < SEQ);
    padd[t] = 0.0f;
    if (okk[t] && pmask[b * SEQ + key] == 0) padd[t] = MASKV;
  }

#pragma unroll
  for (int r = 0; r < 8; ++r) {
    int q = q0 + wave * 16 + r + 8 * g;
    float vals[16]; float mx = -3.0e38f;
#pragma unroll
    for (int t = 0; t < 16; ++t) {
      int key = k0 + 16 * t + lane15;
      float s = acc[t][r] * 0.125f + padd[t];   // 1/sqrt(64)
      int ad = q - key; ad = (ad < 0) ? -ad : ad;
      if (ad > 64) s += MASKV;                  // sliding-window mask
      if (!okk[t]) s = -3.0e38f;                // key does not exist
      vals[t] = s;
      mx = fmaxf(mx, s);
    }
#pragma unroll
    for (int msk = 1; msk < 16; msk <<= 1) mx = fmaxf(mx, __shfl_xor(mx, msk, 32));
    float sum = 0.0f;
#pragma unroll
    for (int t = 0; t < 16; ++t) { float p = __expf(vals[t] - mx); vals[t] = p; sum += p; }
#pragma unroll
    for (int msk = 1; msk < 16; msk <<= 1) sum += __shfl_xor(sum, msk, 32);
    float inv = 1.0f / sum;
    int prow = wave * 16 + r + 8 * g;
#pragma unroll
    for (int t = 0; t < 16; ++t)
      sP[prow * 264 + 16 * t + lane15] = f2bf(vals[t] * inv);
  }
  __syncthreads();

  // out = P * V : 16 rows x 64 d per wave, K = 256 keys
  v8f oac[4];
#pragma unroll
  for (int j = 0; j < 4; ++j) oac[j] = fzero8();
#pragma unroll
  for (int kt = 0; kt < 256; kt += 32) {
    v16bf ap = ld_frag(sP, 264, wave * 16, kt);
#pragma unroll
    for (int j = 0; j < 4; ++j) {
      v16bf bv = ld_frag(sVt, 264, 16 * j, kt);
      oac[j] = wmma_bf16(ap, bv, oac[j]);
    }
  }
#pragma unroll
  for (int j = 0; j < 4; ++j)
#pragma unroll
    for (int r = 0; r < 8; ++r) {
      int row = q0 + wave * 16 + r + 8 * g;
      int d = 16 * j + lane15;
      ob[base + (size_t)row * DM + d] = f2bf(oac[j][r]);
    }
}

// ---------------------------------------------------------------------------
extern "C" void kernel_launch(void* const* d_in, const int* in_sizes, int n_in,
                              void* d_out, int out_size, void* d_ws, size_t ws_size,
                              hipStream_t stream) {
  const float* x      = (const float*)d_in[0];
  const int*   pmask  = (const int*)d_in[1];
  const float* w_qkv  = (const float*)d_in[2];
  const float* w_o    = (const float*)d_in[3];
  const float* g_attn = (const float*)d_in[4];
  const float* g_mlp  = (const float*)d_in[5];
  const float* w_i0   = (const float*)d_in[6];
  const float* w_i1   = (const float*)d_in[7];
  const float* w_mo   = (const float*)d_in[8];
  float* out = (float*)d_out;

  char* ws = (char*)d_ws;
  size_t off = 0;
  auto alloc = [&](size_t bytes) -> char* {
    char* p = ws + off;
    off += (bytes + 255) & ~(size_t)255;
    return p;
  };
  __bf16* h     = (__bf16*)alloc((size_t)TOK * DM * 2);
  __bf16* wqkvT = (__bf16*)alloc((size_t)DM * 3 * DM * 2);
  __bf16* woT   = (__bf16*)alloc((size_t)DM * DM * 2);
  __bf16* wi0T  = (__bf16*)alloc((size_t)DM * DI * 2);
  __bf16* wi1T  = (__bf16*)alloc((size_t)DM * DI * 2);
  __bf16* wmoT  = (__bf16*)alloc((size_t)DI * DM * 2);
  __bf16* qbuf  = (__bf16*)alloc((size_t)TOK * DM * 2);
  __bf16* kbuf  = (__bf16*)alloc((size_t)TOK * DM * 2);
  __bf16* vbuf  = (__bf16*)alloc((size_t)TOK * DM * 2);
  __bf16* attn  = (__bf16*)alloc((size_t)TOK * DM * 2);
  float*  x2    = (float*)alloc((size_t)TOK * DM * 4);
  __bf16* h2    = (__bf16*)alloc((size_t)TOK * DM * 2);
  __bf16* inter = (__bf16*)alloc((size_t)TOK * DI * 2);

  dim3 blk(256);
  auto cdiv = [](size_t a, size_t b) { return (unsigned)((a + b - 1) / b); };

  wtrans_kernel<<<cdiv((size_t)DM * 3 * DM, 256), blk, 0, stream>>>(w_qkv, wqkvT, DM, 3 * DM);
  wtrans_kernel<<<cdiv((size_t)DM * DM, 256), blk, 0, stream>>>(w_o, woT, DM, DM);
  wtrans_kernel<<<cdiv((size_t)DM * DI, 256), blk, 0, stream>>>(w_i0, wi0T, DM, DI);
  wtrans_kernel<<<cdiv((size_t)DM * DI, 256), blk, 0, stream>>>(w_i1, wi1T, DM, DI);
  wtrans_kernel<<<cdiv((size_t)DI * DM, 256), blk, 0, stream>>>(w_mo, wmoT, DI, DM);

  rmsnorm_kernel<<<TOK, blk, 0, stream>>>(x, g_attn, h);

  gemm_kernel<0><<<dim3(3 * DM / 128, TOK / 128), blk, 0, stream>>>(
      h, wqkvT, nullptr, nullptr, nullptr, qbuf, kbuf, vbuf, nullptr, 3 * DM, DM);

  const size_t asmem = (size_t)(128 * 72 + 256 * 72 + 64 * 264 + 128 * 264) * 2;
  (void)hipFuncSetAttribute(reinterpret_cast<const void*>(attn_kernel),
                            hipFuncAttributeMaxDynamicSharedMemorySize, (int)asmem);
  attn_kernel<<<(SEQ / 128) * NH * 2, blk, asmem, stream>>>(qbuf, kbuf, vbuf, pmask, attn);

  gemm_kernel<1><<<dim3(DM / 128, TOK / 128), blk, 0, stream>>>(
      attn, woT, nullptr, x, x2, nullptr, nullptr, nullptr, nullptr, DM, DM);

  rmsnorm_kernel<<<TOK, blk, 0, stream>>>(x2, g_mlp, h2);

  gemm_kernel<2><<<dim3(DI / 128, TOK / 128), blk, 0, stream>>>(
      h2, wi0T, wi1T, nullptr, nullptr, nullptr, nullptr, nullptr, inter, DI, DM);

  gemm_kernel<1><<<dim3(DM / 128, TOK / 128), blk, 0, stream>>>(
      inter, wmoT, nullptr, x2, out, nullptr, nullptr, nullptr, nullptr, DM, DI);

  (void)in_sizes; (void)n_in; (void)out_size; (void)ws_size;
}
